// ADCLayer_32933809225817
// MI455X (gfx1250) — matrix operand order repaired and checked by reading, not compiled, not run on hardware
//
#include <hip/hip_runtime.h>

// ---------------------------------------------------------------------------
// ADCLayer (GAT-style) for MI455X / gfx1250, wave32, BF16 WMMA 16x16x32.
//   * Xw=X@Wv+bv folded into two projection vectors (never materialized).
//   * Only X1 = T@X and X2 = T@X1 live; H = relu([X|X1|X2]@[Wk]+sum bk).
//   * All GEMM operands kept in both orientations so that BOTH A and B tiles
//     stage as pure global_load_async_to_lds_b128 copies. Triple-buffered
//     async pipeline: one full stage always in flight (asynccnt<=8 waits,
//     in-order completion per ISA 4.1). GEMM1/2 write outputs transposed
//     (contiguous per lane); tiny LDS-tile transpose kernels produce the
//     row-major copies the H-GEMM needs.
// ---------------------------------------------------------------------------

typedef __attribute__((ext_vector_type(16))) __bf16 v16bf;
typedef __attribute__((ext_vector_type(8)))  __bf16 v8bf;
typedef __attribute__((ext_vector_type(8)))  float  v8f;
typedef __attribute__((ext_vector_type(4)))  int    v4i;

#define BB   4
#define NN   2048
#define DD   512
#define KCAT (3 * DD)   // 1536

#if __has_builtin(__builtin_amdgcn_global_load_async_to_lds_b128)
#define HAVE_ASYNC 1
#else
#define HAVE_ASYNC 0
#endif

template<int N>
__device__ __forceinline__ void wait_async_le() {
#if HAVE_ASYNC
#if __has_builtin(__builtin_amdgcn_s_wait_asynccnt)
    __builtin_amdgcn_s_wait_asynccnt(N);
#else
    asm volatile("s_wait_asynccnt %0" :: "i"(N) : "memory");
#endif
#endif
}

__device__ __forceinline__ void async_copy_b128(const unsigned short* src, unsigned short* dst) {
#if HAVE_ASYNC
    __builtin_amdgcn_global_load_async_to_lds_b128(
        (__attribute__((address_space(1))) v4i*)src,
        (__attribute__((address_space(3))) v4i*)dst, 0, 0);
#else
    *(uint4*)dst = *(const uint4*)src;
#endif
}

__device__ __forceinline__ unsigned int f32_to_bf16(float f) {
    unsigned int u = __float_as_uint(f);
    u += 0x7FFFu + ((u >> 16) & 1u);      // round-to-nearest-even
    return u >> 16;
}
__device__ __forceinline__ unsigned short to_bf16(float f) { return (unsigned short)f32_to_bf16(f); }
__device__ __forceinline__ unsigned short to_bf16(unsigned short u) { return u; }

// ---------------------------------------------------------------------------
// Kernel 1: uvj/uvi = Wv @ aw halves, cj/ci = bv . aw halves, bsum = sum_k bk
// ---------------------------------------------------------------------------
__global__ void adc_prep(const float* __restrict__ Wv, const float* __restrict__ bv,
                         const float* __restrict__ aw, const float* __restrict__ bk,
                         float* __restrict__ uv, float* __restrict__ cvals,
                         float* __restrict__ bsum) {
    int t = blockIdx.x * blockDim.x + threadIdx.x;
    if (t < DD) {
        const float* row = Wv + (size_t)t * DD;
        float sj = 0.f, si = 0.f;
        for (int d = 0; d < DD; ++d) { float w = row[d]; sj += w * aw[d]; si += w * aw[DD + d]; }
        uv[t] = sj; uv[DD + t] = si;
        bsum[t] = bk[t] + bk[DD + t] + bk[2 * DD + t];
    }
    if (t == 0) {
        float cj = 0.f, ci = 0.f;
        for (int d = 0; d < DD; ++d) { cj += bv[d] * aw[d]; ci += bv[d] * aw[DD + d]; }
        cvals[0] = cj; cvals[1] = ci;
    }
}

// ---------------------------------------------------------------------------
// Kernel 2: tiled transpose (R x C) -> (C x R), output bf16. 32x32 LDS tiles.
// ---------------------------------------------------------------------------
template<typename SrcT>
__global__ void adc_transpose(const SrcT* __restrict__ src, long sSrc, int ldS,
                              unsigned short* __restrict__ dst, long sDst, int ldD) {
    __shared__ unsigned short tile[32][34];
    const int b  = blockIdx.z;
    const int r0 = blockIdx.y * 32, c0 = blockIdx.x * 32;
    const int x  = threadIdx.x & 31, y = threadIdx.x >> 5;   // 32 x 8 threads
#pragma unroll
    for (int yy = y; yy < 32; yy += 8)
        tile[yy][x] = to_bf16(src[(size_t)b * sSrc + (size_t)(r0 + yy) * ldS + c0 + x]);
    __syncthreads();
#pragma unroll
    for (int yy = y; yy < 32; yy += 8)
        dst[(size_t)b * sDst + (size_t)(c0 + yy) * ldD + r0 + x] = tile[x][yy];
}

// ---------------------------------------------------------------------------
// Kernel 3: per row (b,n): sj = X.uvj + cj, si = X.uvi + ci; X -> bf16 slice 0
// ---------------------------------------------------------------------------
__global__ void adc_rowstat(const float* __restrict__ X, const float* __restrict__ uv,
                            const float* __restrict__ cvals,
                            float* __restrict__ sjArr, float* __restrict__ siArr,
                            unsigned short* __restrict__ Xcat) {
    const int row  = blockIdx.x * 8 + (threadIdx.x >> 5);   // [0, B*N)
    const int lane = threadIdx.x & 31;
    const float* x  = X  + (size_t)row * DD + lane * 16;
    const float* uj = uv + lane * 16;
    const float* ui = uv + DD + lane * 16;
    unsigned short* xo = Xcat + (size_t)row * KCAT + lane * 16;
    float sj = 0.f, si = 0.f;
#pragma unroll
    for (int c = 0; c < 4; ++c) {
        float4 v = *(const float4*)(x + c * 4);
        float4 a = *(const float4*)(uj + c * 4);
        float4 b = *(const float4*)(ui + c * 4);
        sj += v.x * a.x + v.y * a.y + v.z * a.z + v.w * a.w;
        si += v.x * b.x + v.y * b.y + v.z * b.z + v.w * b.w;
        *(unsigned int*)(xo + c * 4)     = f32_to_bf16(v.x) | (f32_to_bf16(v.y) << 16);
        *(unsigned int*)(xo + c * 4 + 2) = f32_to_bf16(v.z) | (f32_to_bf16(v.w) << 16);
    }
#pragma unroll
    for (int off = 16; off; off >>= 1) {
        sj += __shfl_down(sj, off, 32);
        si += __shfl_down(si, off, 32);
    }
    if (lane == 0) { sjArr[row] = sj + cvals[0]; siArr[row] = si + cvals[1]; }
}

// ---------------------------------------------------------------------------
// Kernel 4: transition row: t = A*exp(lrelu(sj+si+ab)) / rowsum, stored bf16
// ---------------------------------------------------------------------------
__global__ void adc_trans(const float* __restrict__ A, const float* __restrict__ sjArr,
                          const float* __restrict__ siArr, const float* __restrict__ ab,
                          unsigned short* __restrict__ Tbf) {
    const int b = blockIdx.y, i = blockIdx.x, t = threadIdx.x;
    const float* arow = A + ((size_t)b * NN + i) * NN;
    const float* sj   = sjArr + (size_t)b * NN;
    const float  si   = siArr[(size_t)b * NN + i] + ab[0];
    float w[8]; float part = 0.f;
#pragma unroll
    for (int c = 0; c < 2; ++c) {
        int j = 4 * t + c * 1024;
        float4 s4 = *(const float4*)(sj + j);
        float4 a4 = *(const float4*)(arow + j);
        float e0 = s4.x + si; e0 = e0 > 0.f ? e0 : 0.2f * e0;
        float e1 = s4.y + si; e1 = e1 > 0.f ? e1 : 0.2f * e1;
        float e2 = s4.z + si; e2 = e2 > 0.f ? e2 : 0.2f * e2;
        float e3 = s4.w + si; e3 = e3 > 0.f ? e3 : 0.2f * e3;
        w[4*c+0] = a4.x * __expf(e0);
        w[4*c+1] = a4.y * __expf(e1);
        w[4*c+2] = a4.z * __expf(e2);
        w[4*c+3] = a4.w * __expf(e3);
        part += w[4*c+0] + w[4*c+1] + w[4*c+2] + w[4*c+3];
    }
    __shared__ float red[256];
    red[t] = part; __syncthreads();
    for (int s = 128; s; s >>= 1) { if (t < s) red[t] += red[t + s]; __syncthreads(); }
    const float inv = 1.f / (red[0] + 1e-12f);
    unsigned short* trow = Tbf + ((size_t)b * NN + i) * NN;
#pragma unroll
    for (int c = 0; c < 2; ++c) {
        int j = 4 * t + c * 1024;
        uint2 p;
        p.x = f32_to_bf16(w[4*c+0] * inv) | (f32_to_bf16(w[4*c+1] * inv) << 16);
        p.y = f32_to_bf16(w[4*c+2] * inv) | (f32_to_bf16(w[4*c+3] * inv) << 16);
        *(uint2*)(trow + j) = p;
    }
}

// ---------------------------------------------------------------------------
// Kernel 5: bf16 GEMM  C(MxNc) = A(MxKd) @ B(KdxNc), B supplied transposed.
// 128 threads = 4 waves; block tile 128x128; wave tile 64x64 (16 WMMAs);
// kTile 32; triple-buffered async global->LDS pipeline.
// Fragment layouts per CDNA5 ISA 7.12.2.
// ---------------------------------------------------------------------------
#define TM 128
#define TN 128
#define TK 32
#define LPAD 40                      // padded LDS row (ushorts): 80B, 16B aligned
#define STG  ((TM + TN) * LPAD)      // one pipeline stage (A tile + B tile)

template<int LDA, int LDBT, int LDC, bool OUTF32, bool RELU>
__global__ __launch_bounds__(128)
void adc_gemm(const unsigned short* __restrict__ Ag, long sA,
              const unsigned short* __restrict__ BTg, long sBT,
              unsigned short* __restrict__ CT, float* __restrict__ Cf,
              long sC, const float* __restrict__ bias, int Kd) {
    __shared__ __align__(16) unsigned short SM[3 * STG];

    const int b    = blockIdx.z;
    const int row0 = blockIdx.y * TM;
    const int col0 = blockIdx.x * TN;
    const int t    = threadIdx.x;
    const int lane = t & 31;
    const int wave = t >> 5;      // 0..3
    const int wm   = wave & 1;    // 2 waves along M (64 rows)
    const int wn   = wave >> 1;   // 2 waves along N (64 cols)

    // Staging: thread t owns one 64B row chunk of each tile.
    const unsigned short* aSrc = Ag  + (size_t)b * sA  + (size_t)(row0 + t) * LDA;
    const unsigned short* bSrc = BTg + (size_t)b * sBT + (size_t)(col0 + t) * LDBT;
    const int aOff = t * LPAD;                 // A dst within a stage
    const int bOff = TM * LPAD + t * LPAD;     // B dst within a stage

    auto stage = [&](int soff) {
#pragma unroll
        for (int c = 0; c < 4; ++c) async_copy_b128(aSrc + c * 8, &SM[soff + aOff + c * 8]);
#pragma unroll
        for (int c = 0; c < 4; ++c) async_copy_b128(bSrc + c * 8, &SM[soff + bOff + c * 8]);
        aSrc += TK;
        bSrc += TK;
    };

    v8f acc[4][4];
#pragma unroll
    for (int i = 0; i < 4; ++i)
#pragma unroll
        for (int j = 0; j < 4; ++j)
#pragma unroll
            for (int r = 0; r < 8; ++r) acc[i][j][r] = 0.f;

    // ---- prologue: stages 0 and 1 in flight; wait for stage 0 only ----
    stage(0);
    stage(STG);
    wait_async_le<8>();
    __syncthreads();

    const int am = lane & 15;
    const int kh = lane >> 4;
    const int nk = Kd / TK;

    int cur = 0;        // stage being consumed
    int nxt = 2 * STG;  // stage being filled
    for (int kt = 0; kt < nk; ++kt) {
        const bool more2 = (kt + 2) < nk;
        if (more2) stage(nxt);          // keep one full stage permanently in flight

        union Frag { v16bf v; v8bf h[2]; };
        Frag afr[4], bfr[4];
#pragma unroll
        for (int ms = 0; ms < 4; ++ms) {
            const unsigned short* base = &SM[cur + (wm * 64 + ms * 16 + am) * LPAD];
            afr[ms].h[0] = *(const v8bf*)(base + kh * 8);
            afr[ms].h[1] = *(const v8bf*)(base + 16 + kh * 8);
        }
#pragma unroll
        for (int ns = 0; ns < 4; ++ns) {
            const unsigned short* base = &SM[cur + TM * LPAD + (wn * 64 + ns * 16 + am) * LPAD];
            bfr[ns].h[0] = *(const v8bf*)(base + kh * 8);
            bfr[ns].h[1] = *(const v8bf*)(base + 16 + kh * 8);
        }
#pragma unroll
        for (int ms = 0; ms < 4; ++ms)
#pragma unroll
            for (int ns = 0; ns < 4; ++ns)
                acc[ms][ns] = __builtin_amdgcn_wmma_f32_16x16x32_bf16(
                    false, afr[ms].v, false, bfr[ns].v,
                    (short)0, acc[ms][ns], false, false);

        // async loads complete in order: <=8 outstanding means stage kt+1 done
        if (more2) wait_async_le<8>(); else wait_async_le<0>();
        __syncthreads();

        cur += STG; if (cur == 3 * STG) cur = 0;
        nxt += STG; if (nxt == 3 * STG) nxt = 0;
    }

    // ---- epilogue ----
    const int cn = lane & 15;
    const int cm = (lane >> 4) * 8;
    if (OUTF32) {
        float* out = Cf + (size_t)b * sC;
#pragma unroll
        for (int ms = 0; ms < 4; ++ms)
#pragma unroll
            for (int ns = 0; ns < 4; ++ns) {
                const int col = col0 + wn * 64 + ns * 16 + cn;
                const float bval = bias ? bias[col] : 0.f;
#pragma unroll
                for (int r = 0; r < 8; ++r) {
                    const int rowi = row0 + wm * 64 + ms * 16 + cm + r;
                    float v = acc[ms][ns][r] + bval;
                    if (RELU) v = v > 0.f ? v : 0.f;
                    out[(size_t)rowi * LDC + col] = v;
                }
            }
    } else {
        // transposed bf16 output: CT[col][row], 8 consecutive rows per lane
        unsigned short* out = CT + (size_t)b * sC;
#pragma unroll
        for (int ms = 0; ms < 4; ++ms)
#pragma unroll
            for (int ns = 0; ns < 4; ++ns) {
                const int col  = col0 + wn * 64 + ns * 16 + cn;
                const int rowb = row0 + wm * 64 + ms * 16 + cm;
                uint4 p;
                p.x = f32_to_bf16(acc[ms][ns][0]) | (f32_to_bf16(acc[ms][ns][1]) << 16);
                p.y = f32_to_bf16(acc[ms][ns][2]) | (f32_to_bf16(acc[ms][ns][3]) << 16);
                p.z = f32_to_bf16(acc[ms][ns][4]) | (f32_to_bf16(acc[ms][ns][5]) << 16);
                p.w = f32_to_bf16(acc[ms][ns][6]) | (f32_to_bf16(acc[ms][ns][7]) << 16);
                *(uint4*)(out + (size_t)col * LDC + rowb) = p;
            }
    }
}

// ---------------------------------------------------------------------------
extern "C" void kernel_launch(void* const* d_in, const int* in_sizes, int n_in,
                              void* d_out, int out_size, void* d_ws, size_t ws_size,
                              hipStream_t stream) {
    const float* X  = (const float*)d_in[0];
    const float* A  = (const float*)d_in[1];
    const float* Wv = (const float*)d_in[2];
    const float* bv = (const float*)d_in[3];
    const float* aw = (const float*)d_in[4];
    const float* ab = (const float*)d_in[5];
    const float* Wk = (const float*)d_in[6];
    const float* bk = (const float*)d_in[7];
    float* out = (float*)d_out;
    (void)in_sizes; (void)n_in; (void)out_size; (void)ws_size;

    char* ws = (char*)d_ws;
    size_t off = 0;
    auto alloc = [&](size_t bytes) -> void* {
        void* p = ws + off;
        off = (off + bytes + 255) & ~(size_t)255;
        return p;
    };
    float* uv    = (float*)alloc(2 * DD * sizeof(float));
    float* cvals = (float*)alloc(2 * sizeof(float));
    float* bsum  = (float*)alloc(DD * sizeof(float));
    float* sj    = (float*)alloc((size_t)BB * NN * sizeof(float));
    float* si    = (float*)alloc((size_t)BB * NN * sizeof(float));
    unsigned short* Xcat = (unsigned short*)alloc((size_t)BB * NN * KCAT * 2); // [X|X1|X2] row-major
    unsigned short* Tbf  = (unsigned short*)alloc((size_t)BB * NN * NN * 2);   // transition
    unsigned short* XT0  = (unsigned short*)alloc((size_t)BB * DD * NN * 2);   // X^T   (512x2048)
    unsigned short* X1T  = (unsigned short*)alloc((size_t)BB * DD * NN * 2);   // X1^T
    unsigned short* X2T  = (unsigned short*)alloc((size_t)BB * DD * NN * 2);   // X2^T
    unsigned short* WT   = (unsigned short*)alloc((size_t)DD * KCAT * 2);      // Wk^T (512x1536)

    const long sT  = (long)NN * NN;
    const long sXc = (long)NN * KCAT;
    const long sXT = (long)DD * NN;

    adc_prep<<<2, 256, 0, stream>>>(Wv, bv, aw, bk, uv, cvals, bsum);
    adc_rowstat<<<BB * NN / 8, 256, 0, stream>>>(X, uv, cvals, sj, si, Xcat);
    adc_trans<<<dim3(NN, BB), 256, 0, stream>>>(A, sj, si, ab, Tbf);
    // X^T  (per batch 2048x512 -> 512x2048), Wk^T (1536x512 -> 512x1536)
    adc_transpose<float><<<dim3(DD / 32, NN / 32, BB), 256, 0, stream>>>(
        X, (long)NN * DD, DD, XT0, sXT, NN);
    adc_transpose<float><<<dim3(DD / 32, KCAT / 32, 1), 256, 0, stream>>>(
        Wk, 0, DD, WT, 0, KCAT);

    dim3 g(DD / TN, NN / TM, BB);   // (4, 16, 4)
    // X1^T = (T @ X)^T
    adc_gemm<NN, NN, NN, false, false><<<g, 128, 0, stream>>>(
        Tbf, sT, XT0, sXT, X1T, nullptr, sXT, nullptr, NN);
    // Xcat slice1 = X1 (row-major) from X1^T
    adc_transpose<unsigned short><<<dim3(NN / 32, DD / 32, BB), 256, 0, stream>>>(
        X1T, sXT, NN, Xcat + DD, sXc, KCAT);
    // X2^T = (T @ X1)^T
    adc_gemm<NN, NN, NN, false, false><<<g, 128, 0, stream>>>(
        Tbf, sT, X1T, sXT, X2T, nullptr, sXT, nullptr, NN);
    // Xcat slice2 = X2 (row-major) from X2^T
    adc_transpose<unsigned short><<<dim3(NN / 32, DD / 32, BB), 256, 0, stream>>>(
        X2T, sXT, NN, Xcat + 2 * DD, sXc, KCAT);
    // H = relu([X|X1|X2] @ Wcat + bsum) -> d_out (f32 row-major)
    adc_gemm<KCAT, KCAT, DD, true, true><<<g, 128, 0, stream>>>(
        Xcat, sXc, WT, 0, nullptr, out, (long)NN * DD, bsum, KCAT);
}
